// NeighborhoodAttention_40836549050735
// MI455X (gfx1250) — matrix operand order repaired and checked by reading, not compiled, not run on hardware
//
#include <hip/hip_runtime.h>
#include <hip/hip_bf16.h>

#define NTOK   (8*48*48)   // 18432 tokens
#define TT     8
#define HH     48
#define WW     48
#define HEADS  8
#define HD     64
#define EMBED  512

typedef __bf16 bf16x16 __attribute__((ext_vector_type(16)));
typedef __bf16 bf16x8  __attribute__((ext_vector_type(8)));
typedef float  f32x8   __attribute__((ext_vector_type(8)));
typedef int    v4i     __attribute__((ext_vector_type(4)));
typedef unsigned short u16;
typedef unsigned int   u32;

#if defined(__has_builtin)
#if __has_builtin(__builtin_amdgcn_global_load_async_to_lds_b128)
#define HAVE_ASYNC_LDS 1
#endif
#if __has_builtin(__builtin_amdgcn_s_wait_asynccnt)
#define HAVE_WAIT_ASYNC 1
#endif
#endif

// builtin signature (from hipcc diagnostic): (v4i addrspace(1)*, v4i addrspace(3)*, imm, imm)
typedef __attribute__((address_space(1))) v4i* as1_v4i_p;
typedef __attribute__((address_space(3))) v4i* as3_v4i_p;

__device__ __forceinline__ u16 f2bf(float f) {
  u32 u = __builtin_bit_cast(u32, f);
  u += 0x7FFFu + ((u >> 16) & 1u);
  return (u16)(u >> 16);
}
__device__ __forceinline__ float bf2f(u16 h) {
  u32 u = ((u32)h) << 16;
  return __builtin_bit_cast(float, u);
}
__device__ __forceinline__ bf16x16 cat16(bf16x8 lo, bf16x8 hi) {
  return __builtin_shufflevector(lo, hi, 0,1,2,3,4,5,6,7,8,9,10,11,12,13,14,15);
}

// 16-byte memory -> LDS copy; async (ASYNCcnt / GLOBAL_LOAD_ASYNC_TO_LDS_B128)
// when available, else synchronous via VGPRs.
__device__ __forceinline__ void copy16_to_lds(void* lds_dst, const void* gsrc) {
#ifdef HAVE_ASYNC_LDS
  __builtin_amdgcn_global_load_async_to_lds_b128(
      (as1_v4i_p)(v4i*)gsrc, (as3_v4i_p)lds_dst, 0, 0);
#else
  *(uint4*)lds_dst = *(const uint4*)gsrc;
#endif
}
__device__ __forceinline__ void wait_async_copies() {
#ifdef HAVE_ASYNC_LDS
#ifdef HAVE_WAIT_ASYNC
  __builtin_amdgcn_s_wait_asynccnt(0);
#else
  asm volatile("s_wait_asynccnt 0" ::: "memory");
#endif
#endif
}

// ---------------------------------------------------------------- conversions
__global__ void cvt_f32_to_bf16(const float* __restrict__ src,
                                u16* __restrict__ dst, int n) {
  int i = blockIdx.x * blockDim.x + threadIdx.x;
  if (i < n) dst[i] = f2bf(src[i]);
}

// ---------------------------------------------------------------- NT GEMM via WMMA
// C[m,n] = sum_k A[m,k] * B[n,k]   (A: MxK bf16 row-major, B: NxK bf16 row-major)
// One wave computes a 16x64 strip: A fragment reused across 4 B tiles.
template<bool OUT_BF16, int K>
__global__ __launch_bounds__(32)
void gemm_nt_wmma(const u16* __restrict__ A, const u16* __restrict__ Bw,
                  void* __restrict__ C, int M, int N) {
  const int lane = threadIdx.x;
  const int col  = lane & 15;
  const int sel  = lane >> 4;
  const int n0 = blockIdx.x * 64;
  const int m0 = blockIdx.y * 16;
  const u16* arow  = A  + (size_t)(m0 + col) * K;
  const u16* brow0 = Bw + (size_t)(n0 + col) * K;
  f32x8 acc[4] = {};
#pragma unroll 4
  for (int kb = 0; kb < K; kb += 32) {
    // A fragment: 16 contiguous K per lane-half (matches B)
    bf16x8 a0 = *(const bf16x8*)(arow + kb + sel * 16);
    bf16x8 a1 = *(const bf16x8*)(arow + kb + sel * 16 + 8);
    bf16x16 af = cat16(a0, a1);
#pragma unroll
    for (int nt = 0; nt < 4; ++nt) {
      // B fragment: same order as A -> 16 contiguous K per lane-half
      const u16* br = brow0 + (size_t)nt * 16 * K;
      bf16x8 b0 = *(const bf16x8*)(br + kb + sel * 16);
      bf16x8 b1 = *(const bf16x8*)(br + kb + sel * 16 + 8);
      acc[nt] = __builtin_amdgcn_wmma_f32_16x16x32_bf16(
          false, af, false, cat16(b0, b1), (short)0, acc[nt], false, false);
    }
  }
#pragma unroll
  for (int nt = 0; nt < 4; ++nt)
#pragma unroll
    for (int r = 0; r < 8; ++r) {
      int m = m0 + r + 8 * sel;
      int n = n0 + nt * 16 + col;
      if (OUT_BF16) ((u16*)C)[(size_t)m * N + n] = f2bf(acc[nt][r]);
      else          ((float*)C)[(size_t)m * N + n] = acc[nt][r];
    }
}

// ---------------------------------------------------------------- RMSNorm + axial RoPE
__global__ void rmsrope_kernel(const u16* __restrict__ qkv,
                               const float* __restrict__ qw,
                               const float* __restrict__ kw,
                               u16* __restrict__ qo, u16* __restrict__ ko) {
  int gid = blockIdx.x * blockDim.x + threadIdx.x;
  if (gid >= NTOK * HEADS) return;
  int head = gid & 7;
  int tok  = gid >> 3;
  int t   = tok / (HH * WW);
  int rem = tok % (HH * WW);
  int xr = rem / WW, yr = rem % WW;

  // axial rope phases: 30 pairs = 3 axes x 10 freqs (linspace(1,128,10)*pi)
  const float PI = 3.14159265358979323846f;
  float pt = -1.f + 2.f * (float)t  / (float)(TT - 1);
  float ph = -1.f + 2.f * (float)xr / (float)(HH - 1);
  float pw = -1.f + 2.f * (float)yr / (float)(WW - 1);
  float cs[30], sn[30];
#pragma unroll
  for (int i = 0; i < 30; ++i) {
    int axis = i / 10, idx = i % 10;
    float pos = (axis == 0) ? pt : ((axis == 1) ? ph : pw);
    float f = pos * ((1.f + 127.f * (float)idx / 9.f) * PI);
    cs[i] = __cosf(f);
    sn[i] = __sinf(f);
  }

  const u16* base = qkv + (size_t)tok * (3 * EMBED) + head * 192;
  for (int pass = 0; pass < 2; ++pass) {
    const u16*  src = base + pass * 64;
    const float* wv = pass ? kw : qw;
    u16* dst = (pass ? ko : qo) + ((size_t)tok * HEADS + head) * HD;
    float v[HD];
    float ss = 0.f;
#pragma unroll
    for (int d = 0; d < HD; ++d) { v[d] = bf2f(src[d]); ss += v[d] * v[d]; }
    float r = rsqrtf(ss * (1.f / HD) + 1e-6f);
#pragma unroll
    for (int d = 0; d < HD; ++d) v[d] *= r * wv[d];
#pragma unroll
    for (int i = 0; i < 30; ++i) {
      float v0 = v[2 * i], v1 = v[2 * i + 1];
      v[2 * i]     = v0 * cs[i] - v1 * sn[i];
      v[2 * i + 1] = v1 * cs[i] + v0 * sn[i];
    }
#pragma unroll
    for (int d = 0; d < HD; ++d) dst[d] = f2bf(v[d]);
  }
}

// ---------------------------------------------------------------- neighborhood attention
// One wave per (x,y,head). All 8 time-queries at (x,y) share one 72-key window.
__global__ __launch_bounds__(32)
void attn_kernel(const u16* __restrict__ Q, const u16* __restrict__ Kn,
                 const u16* __restrict__ QKV, u16* __restrict__ O) {
  __shared__ __attribute__((aligned(16))) u16  Qs[16 * 64];   // 8 real rows + 8 zero
  __shared__ __attribute__((aligned(16))) u16  Ks[80 * 64];   // 72 keys + 8 zero
  __shared__ __attribute__((aligned(16))) u16  Vs[96 * 64];   // 72 vals + 24 zero
  __shared__ __attribute__((aligned(16))) float Ss[16 * 80];  // scores
  __shared__ __attribute__((aligned(16))) u16  Ps[16 * 96];   // probs (zero padded)

  const int lane = threadIdx.x;
  const int yy = blockIdx.x, xx = blockIdx.y, head = blockIdx.z;
  int x0 = xx - 1; x0 = x0 < 0 ? 0 : (x0 > HH - 3 ? HH - 3 : x0);
  int y0 = yy - 1; y0 = y0 < 0 ? 0 : (y0 > WW - 3 ? WW - 3 : y0);

  // stage Q rows t=0..7 (8 rows x 128B = 64 x 16B chunks), async mem->LDS
  for (int i = lane; i < 64; i += 32) {
    int t = i >> 3, seg = i & 7;
    copy16_to_lds(Qs + t * 64 + seg * 8,
                  Q + ((((size_t)t * HH + xx) * WW + yy) * HEADS + head) * HD + seg * 8);
  }
  // stage K and V windows: 72 rows x 8 chunks each
  for (int i = lane; i < 72 * 8; i += 32) {
    int wr = i >> 3, seg = i & 7;
    int s = wr / 9, rr = wr % 9;
    int kx = x0 + rr / 3, ky = y0 + rr % 3;
    size_t posoff = ((size_t)s * HH + kx) * WW + ky;
    copy16_to_lds(Ks + wr * 64 + seg * 8,
                  Kn + (posoff * HEADS + head) * HD + seg * 8);
    copy16_to_lds(Vs + wr * 64 + seg * 8,
                  QKV + posoff * (3 * EMBED) + head * 192 + 128 + seg * 8);
  }
  // zero padding rows (plain LDS stores; DS ops are in-order within the wave)
  for (int i = lane; i < 8 * 32; i += 32)  ((u32*)Qs)[8 * 32 + i] = 0u;
  for (int i = lane; i < 8 * 32; i += 32)  ((u32*)Ks)[72 * 32 + i] = 0u;
  for (int i = lane; i < 24 * 32; i += 32) ((u32*)Vs)[72 * 32 + i] = 0u;
  wait_async_copies();
  __syncthreads();

  const int col = lane & 15;
  const int sel = lane >> 4;

  // ---- scores = Q (16x64) @ K^T (64x80): 2 k-steps x 5 n-tiles
  f32x8 acc[5] = {};
#pragma unroll
  for (int kb = 0; kb < 64; kb += 32) {
    bf16x8 a0 = *(const bf16x8*)(Qs + col * 64 + kb + sel * 16);
    bf16x8 a1 = *(const bf16x8*)(Qs + col * 64 + kb + sel * 16 + 8);
    bf16x16 af = cat16(a0, a1);
#pragma unroll
    for (int nt = 0; nt < 5; ++nt) {
      bf16x8 b0 = *(const bf16x8*)(Ks + (nt * 16 + col) * 64 + kb + sel * 16);
      bf16x8 b1 = *(const bf16x8*)(Ks + (nt * 16 + col) * 64 + kb + sel * 16 + 8);
      acc[nt] = __builtin_amdgcn_wmma_f32_16x16x32_bf16(
          false, af, false, cat16(b0, b1), (short)0, acc[nt], false, false);
    }
  }
#pragma unroll
  for (int nt = 0; nt < 5; ++nt)
#pragma unroll
    for (int r = 0; r < 8; ++r)
      Ss[(r + 8 * sel) * 80 + nt * 16 + col] = acc[nt][r] * 0.125f;  // 1/sqrt(64)
  __syncthreads();

  // ---- softmax over the 72 valid keys (one row per lane 0..7)
  for (int i = lane; i < 16 * 48; i += 32) ((u32*)Ps)[i] = 0u;  // zero all probs
  __syncthreads();
  if (lane < 8) {
    float mx = -1e30f;
    for (int c = 0; c < 72; ++c) mx = fmaxf(mx, Ss[lane * 80 + c]);
    float sum = 0.f;
    for (int c = 0; c < 72; ++c) sum += __expf(Ss[lane * 80 + c] - mx);
    float inv = 1.f / sum;
    for (int c = 0; c < 72; ++c)
      Ps[lane * 96 + c] = f2bf(__expf(Ss[lane * 80 + c] - mx) * inv);
  }
  __syncthreads();

  // ---- out = P (16x96) @ V (96x64): 3 k-steps x 4 n-tiles
  f32x8 oacc[4] = {};
#pragma unroll
  for (int kb = 0; kb < 96; kb += 32) {
    bf16x8 a0 = *(const bf16x8*)(Ps + col * 96 + kb + sel * 16);
    bf16x8 a1 = *(const bf16x8*)(Ps + col * 96 + kb + sel * 16 + 8);
    bf16x16 af = cat16(a0, a1);
#pragma unroll
    for (int nt = 0; nt < 4; ++nt) {
      int n = nt * 16 + col;
      bf16x16 bfr;
#pragma unroll
      for (int e = 0; e < 16; ++e)
        bfr[e] = __builtin_bit_cast(__bf16, Vs[(kb + sel * 16 + e) * 64 + n]);
      oacc[nt] = __builtin_amdgcn_wmma_f32_16x16x32_bf16(
          false, af, false, bfr, (short)0, oacc[nt], false, false);
    }
  }
  if (sel == 0) {
#pragma unroll
    for (int nt = 0; nt < 4; ++nt)
#pragma unroll
      for (int r = 0; r < 8; ++r) {
        int d = nt * 16 + col;
        O[((((size_t)r * HH + xx) * WW + yy) * HEADS + head) * HD + d] = f2bf(oacc[nt][r]);
      }
  }
}

// ---------------------------------------------------------------- host launcher
extern "C" void kernel_launch(void* const* d_in, const int* in_sizes, int n_in,
                              void* d_out, int out_size, void* d_ws, size_t ws_size,
                              hipStream_t stream) {
  const float* x     = (const float*)d_in[0];   // [NTOK, 512]
  const float* w_qkv = (const float*)d_in[1];   // [1536, 512]
  const float* w_out = (const float*)d_in[2];   // [512, 512]
  const float* qnw   = (const float*)d_in[3];   // [64]
  const float* knw   = (const float*)d_in[4];   // [64]

  char* ws = (char*)d_ws;
  size_t oX   = 0;
  size_t oWq  = oX   + (size_t)NTOK * EMBED * 2;        // x bf16
  size_t oWo  = oWq  + (size_t)3 * EMBED * EMBED * 2;   // w_qkv bf16
  size_t oQKV = oWo  + (size_t)EMBED * EMBED * 2;       // w_out bf16
  size_t oQ   = oQKV + (size_t)NTOK * 3 * EMBED * 2;    // qkv bf16
  size_t oK   = oQ   + (size_t)NTOK * EMBED * 2;        // q roped bf16
  size_t oAO  = oK   + (size_t)NTOK * EMBED * 2;        // k roped bf16
  // total = 128 MB
  u16* x_bf   = (u16*)(ws + oX);
  u16* wq_bf  = (u16*)(ws + oWq);
  u16* wo_bf  = (u16*)(ws + oWo);
  u16* qkv_bf = (u16*)(ws + oQKV);
  u16* q_bf   = (u16*)(ws + oQ);
  u16* k_bf   = (u16*)(ws + oK);
  u16* ao_bf  = (u16*)(ws + oAO);

  int nx = NTOK * EMBED, nwq = 3 * EMBED * EMBED, nwo = EMBED * EMBED;
  cvt_f32_to_bf16<<<(nx  + 255) / 256, 256, 0, stream>>>(x,     x_bf,  nx);
  cvt_f32_to_bf16<<<(nwq + 255) / 256, 256, 0, stream>>>(w_qkv, wq_bf, nwq);
  cvt_f32_to_bf16<<<(nwo + 255) / 256, 256, 0, stream>>>(w_out, wo_bf, nwo);

  // qkv = x @ w_qkv^T  : M=18432, N=1536, K=512
  gemm_nt_wmma<true, EMBED><<<dim3(3 * EMBED / 64, NTOK / 16), 32, 0, stream>>>(
      x_bf, wq_bf, (void*)qkv_bf, NTOK, 3 * EMBED);

  rmsrope_kernel<<<(NTOK * HEADS + 255) / 256, 256, 0, stream>>>(
      qkv_bf, qnw, knw, q_bf, k_bf);

  attn_kernel<<<dim3(WW, HH, HEADS), 32, 0, stream>>>(q_bf, k_bf, qkv_bf, ao_bf);

  // out = attn @ w_out^T : M=18432, N=512, K=512 (fp32 out)
  gemm_nt_wmma<false, EMBED><<<dim3(EMBED / 64, NTOK / 16), 32, 0, stream>>>(
      ao_bf, wo_bf, d_out, NTOK, EMBED);
}
